// SmallSwitchMLP_45844480917645
// MI455X (gfx1250) — matrix operand order repaired and checked by reading, not compiled, not run on hardware
//
#include <hip/hip_runtime.h>

#define HIDDEN 512
#define FFN    2048
#define NEXP   8
#define NTOK   8192
#define TILE_M 32
#define CHUNK  256
#define XLD    528   // padded bf16 row stride for x tile (16B-aligned rows, +8 dword skew)
#define HLD    272   // padded bf16 row stride for h chunk tile

// Packed-fragment geometry: each WMMA B fragment (32K x 16N bf16) = 256 dwords (1KB),
// lane's 8 dwords contiguous at frag*256 + lane*8. k-steps are consecutive fragments.
#define NT1 (FFN / 16)      // 128 n-tiles in W1
#define KS1 (HIDDEN / 32)   // 16 k-steps in W1
#define NT2 (HIDDEN / 16)   // 32 n-tiles in W2
#define KS2 (FFN / 32)      // 64 k-steps in W2
#define FRAGS1 (NT1 * KS1)  // 2048 per expert
#define FRAGS2 (NT2 * KS2)  // 2048 per expert

typedef __attribute__((ext_vector_type(16))) __bf16 v16bf;
typedef __attribute__((ext_vector_type(2)))  __bf16 v2bf;
typedef __attribute__((ext_vector_type(8)))  float  v8f;
typedef __attribute__((ext_vector_type(2)))  float  v2f;

// Native f32->bf16 conversions (RNE) -- let the compiler emit v_cvt_*bf16*.
__device__ __forceinline__ unsigned short bf16_of(float f) {
  __bf16 b = (__bf16)f;
  return __builtin_bit_cast(unsigned short, b);
}
__device__ __forceinline__ unsigned pack2_bf16(float lo, float hi) {
  v2f  f = {lo, hi};
  v2bf b = __builtin_convertvector(f, v2bf);
  return __builtin_bit_cast(unsigned, b);
}

union AFrag { v16bf v; uint4 q[2]; };
union BFrag { v16bf v; uint4 q[2]; unsigned u[8]; };

// A-matrix 16x32 bf16 fragment from LDS (contiguous k order, same as packed B):
// lanes 0-15: M=lane, K = k0..k0+7 (q0) and k0+8..k0+15 (q1)
// lanes 16-31: M=lane-16, K = k0+16..k0+23 (q0) and k0+24..k0+31 (q1)
__device__ __forceinline__ v16bf load_a_lds(const unsigned short* base, int ld,
                                            int m0, int k0, int lane) {
  int row = m0 + (lane & 15);
  int kA  = k0 + (lane & 16);
  AFrag r;
  r.q[0] = *(const uint4*)(base + row * ld + kA);
  r.q[1] = *(const uint4*)(base + row * ld + kA + 8);
  return r.v;
}

__device__ __forceinline__ v8f wmma_bf16(v16bf a, v16bf b, v8f c) {
  return __builtin_amdgcn_wmma_f32_16x16x32_bf16(false, a, false, b, (short)0, c,
                                                 false, false);
}

__global__ void init_kernel(int* __restrict__ cursor) {
  if (threadIdx.x < NEXP) cursor[threadIdx.x] = 0;
}

// Pre-pack a row-major f32 KxN weight tensor [NEXP,K,N] into per-fragment bf16:
// wave per fragment; frag id = e*(ntiles*ksteps) + nt*ksteps + ks.
// Lane n<16: N=nt*16+n, K=ks*32..+15 ; lane n+16: same N, K=ks*32+16..+31.
__global__ void __launch_bounds__(128)
pack_kernel(const float* __restrict__ W, unsigned* __restrict__ out,
            int K, int N, int ksteps) {
  int wid  = (blockIdx.x * blockDim.x + threadIdx.x) >> 5;
  int lane = threadIdx.x & 31;
  int fragsPerExp = (N >> 4) * ksteps;
  int e  = wid / fragsPerExp;
  int f  = wid - e * fragsPerExp;
  int nt = f / ksteps;
  int ks = f - nt * ksteps;

  const float* We = W + (size_t)e * K * N;
  int col = (nt << 4) + (lane & 15);
  int kb  = (ks << 5) + (lane & 16);
  const float* p = We + (size_t)kb * N + col;

  BFrag r;
#pragma unroll
  for (int i = 0; i < 8; ++i) {
    r.u[i] = pack2_bf16(p[0], p[(size_t)N]);
    p += (size_t)2 * N;
  }
  uint4* o = (uint4*)(out + (size_t)wid * 256 + lane * 8);
  o[0] = r.q[0];
  o[1] = r.q[1];
}

// One wave32 per token: logits, softmax, argmax, gate, compaction list.
__global__ void __launch_bounds__(128)
gate_kernel(const float* __restrict__ x, const float* __restrict__ Wg,
            int* __restrict__ cursor, int* __restrict__ idx_buf,
            float* __restrict__ gate_buf, float* __restrict__ probs_buf,
            int* __restrict__ list) {
  int lane  = threadIdx.x & 31;
  int wv    = threadIdx.x >> 5;
  int token = blockIdx.x * 4 + wv;

  float acc[NEXP];
#pragma unroll
  for (int e = 0; e < NEXP; ++e) acc[e] = 0.0f;

  const float* xr = x + (size_t)token * HIDDEN;
  for (int h = lane; h < HIDDEN; h += 32) {
    float xv = xr[h];
    const float* wr = Wg + (size_t)h * NEXP;
#pragma unroll
    for (int e = 0; e < NEXP; ++e) acc[e] += xv * wr[e];
  }
#pragma unroll
  for (int off = 16; off > 0; off >>= 1)
#pragma unroll
    for (int e = 0; e < NEXP; ++e) acc[e] += __shfl_down(acc[e], off, 32);

  if (lane == 0) {
    float mx = acc[0]; int am = 0;
#pragma unroll
    for (int e = 1; e < NEXP; ++e)
      if (acc[e] > mx) { mx = acc[e]; am = e; }   // first-max tie rule like argmax
    float pr[NEXP], sum = 0.0f;
#pragma unroll
    for (int e = 0; e < NEXP; ++e) { pr[e] = __expf(acc[e] - mx); sum += pr[e]; }
    float inv = 1.0f / sum;
#pragma unroll
    for (int e = 0; e < NEXP; ++e) {
      pr[e] *= inv;
      probs_buf[(size_t)token * NEXP + e] = pr[e];
    }
    float s = pr[am];
    idx_buf[token]  = am;
    gate_buf[token] = s / (s + 1e-8f);
    int pos = atomicAdd(&cursor[am], 1);
    list[(size_t)am * NTOK + pos] = token;
  }
}

// Deterministic fixed-tree reduction for the load-balancing loss.
__global__ void __launch_bounds__(256)
loss_kernel(const float* __restrict__ probs_buf, const int* __restrict__ idx_buf,
            float* __restrict__ out_loss) {
  __shared__ float rbuf[256];
  __shared__ float totP[NEXP], totC[NEXP];
  int tid = threadIdx.x;

  float psum[NEXP], csum[NEXP];
#pragma unroll
  for (int e = 0; e < NEXP; ++e) { psum[e] = 0.0f; csum[e] = 0.0f; }
  for (int tk = tid; tk < NTOK; tk += 256) {
    int ix = idx_buf[tk];
#pragma unroll
    for (int e = 0; e < NEXP; ++e) {
      psum[e] += probs_buf[(size_t)tk * NEXP + e];
      csum[e] += (ix == e) ? 1.0f : 0.0f;
    }
  }
  for (int e = 0; e < NEXP; ++e) {
    rbuf[tid] = psum[e]; __syncthreads();
    for (int s = 128; s > 0; s >>= 1) {
      if (tid < s) rbuf[tid] += rbuf[tid + s];
      __syncthreads();
    }
    if (tid == 0) totP[e] = rbuf[0];
    __syncthreads();
    rbuf[tid] = csum[e]; __syncthreads();
    for (int s = 128; s > 0; s >>= 1) {
      if (tid < s) rbuf[tid] += rbuf[tid + s];
      __syncthreads();
    }
    if (tid == 0) totC[e] = rbuf[0];
    __syncthreads();
  }
  if (tid == 0) {
    float lb = 0.0f;
#pragma unroll
    for (int e = 0; e < NEXP; ++e)
      lb += (totP[e] / (float)NTOK) * (totC[e] / (float)NTOK);
    out_loss[0] = (float)NEXP * lb;
  }
}

// Routed, fused two-GEMM expert MLP: 32-token tile, bf16 WMMA with pre-packed
// B fragments streamed from L2, h staged in LDS, 32x512 f32 output in registers.
__global__ void __launch_bounds__(128)
moe_kernel(const float* __restrict__ x,
           const unsigned* __restrict__ W1p, const float* __restrict__ b1,
           const unsigned* __restrict__ W2p, const float* __restrict__ b2,
           const float* __restrict__ gate_buf, const int* __restrict__ list,
           const int* __restrict__ cursor, float* __restrict__ out) {
  const int e     = blockIdx.y;
  const int cnt   = cursor[e];
  const int tile0 = blockIdx.x * TILE_M;
  if (tile0 >= cnt) return;

  const int tid   = threadIdx.x;
  const int lane  = tid & 31;
  const int wave  = tid >> 5;
  const int cl    = lane & 15;
  const int mhalf = (lane >> 4) & 1;   // C/D layout: lanes 16-31 hold M+8

  const float* b1e = b1 + (size_t)e * FFN;
  const float* b2e = b2 + (size_t)e * HIDDEN;
  const int*   le  = list + (size_t)e * NTOK;

  __shared__ __attribute__((aligned(16))) unsigned short xs[TILE_M * XLD];
  __shared__ __attribute__((aligned(16))) unsigned short hs[TILE_M * HLD];
  __shared__ int   tok_s[TILE_M];
  __shared__ float gate_s[TILE_M];

  if (tid < TILE_M) {
    int g = tile0 + tid;
    int t = (g < cnt) ? le[g] : -1;
    tok_s[tid]  = t;
    gate_s[tid] = (t >= 0) ? gate_buf[t] : 0.0f;
  }
  __syncthreads();

  for (int i = tid; i < TILE_M * HIDDEN; i += 128) {
    int row = i >> 9;              // / HIDDEN
    int col = i & (HIDDEN - 1);
    int t = tok_s[row];
    float v = (t >= 0) ? x[(size_t)t * HIDDEN + col] : 0.0f;
    xs[row * XLD + col] = bf16_of(v);
  }
  __syncthreads();

  v8f vzero = {};
  v8f yacc[2][8];
#pragma unroll
  for (int m = 0; m < 2; ++m)
#pragma unroll
    for (int n = 0; n < 8; ++n) yacc[m][n] = vzero;

  for (int c = 0; c < FFN; c += CHUNK) {
    // ---- GEMM1: h[:, c:c+CHUNK] = relu(x @ W1 + b1), bf16 into LDS ----
#pragma unroll 1
    for (int nt = 0; nt < 4; ++nt) {
      int n0 = c + (wave * 4 + nt) * 16;
      float b1v = b1e[n0 + cl];
      // fragment base: (e, ntile = n0/16, ks = 0..KS1-1 consecutive)
      const uint4* bp = (const uint4*)W1p +
                        ((size_t)e * FRAGS1 + (size_t)(n0 >> 4) * KS1) * 64 +
                        lane * 2;
      v8f acc0 = vzero, acc1 = vzero;
#pragma unroll 2
      for (int k0 = 0; k0 < HIDDEN; k0 += 32) {
        __builtin_prefetch(bp + 256, 0, 3);  // 4 fragments (4KB) ahead
        BFrag b;
        b.q[0] = bp[0];
        b.q[1] = bp[1];
        bp += 64;                            // next k-step fragment (1KB)
        v16bf a0 = load_a_lds(xs, XLD, 0,  k0, lane);
        v16bf a1 = load_a_lds(xs, XLD, 16, k0, lane);
        acc0 = wmma_bf16(a0, b.v, acc0);
        acc1 = wmma_bf16(a1, b.v, acc1);
      }
      int hc = (n0 - c) + cl;
#pragma unroll
      for (int r = 0; r < 8; ++r) {
        int row = r + mhalf * 8;
        hs[row * HLD + hc]        = bf16_of(fmaxf(acc0[r] + b1v, 0.0f));
        hs[(row + 16) * HLD + hc] = bf16_of(fmaxf(acc1[r] + b1v, 0.0f));
      }
    }
    __syncthreads();
    // ---- GEMM2 partial: y += h_chunk @ W2[c:c+CHUNK, :] ----
#pragma unroll 1
    for (int nt = 0; nt < 8; ++nt) {
      int n0 = wave * 128 + nt * 16;
      const uint4* bp = (const uint4*)W2p +
                        ((size_t)e * FRAGS2 + (size_t)(n0 >> 4) * KS2 + (c >> 5)) * 64 +
                        lane * 2;
#pragma unroll 2
      for (int k0 = 0; k0 < CHUNK; k0 += 32) {
        __builtin_prefetch(bp + 256, 0, 3);
        BFrag b;
        b.q[0] = bp[0];
        b.q[1] = bp[1];
        bp += 64;
        v16bf a0 = load_a_lds(hs, HLD, 0,  k0, lane);
        v16bf a1 = load_a_lds(hs, HLD, 16, k0, lane);
        yacc[0][nt] = wmma_bf16(a0, b.v, yacc[0][nt]);
        yacc[1][nt] = wmma_bf16(a1, b.v, yacc[1][nt]);
      }
    }
    __syncthreads();
  }

  // ---- epilogue: out[token] = gate * (y + b2) ----
#pragma unroll 1
  for (int nt = 0; nt < 8; ++nt) {
    int col = wave * 128 + nt * 16 + cl;
    float b2v = b2e[col];
#pragma unroll
    for (int r = 0; r < 8; ++r) {
#pragma unroll
      for (int m = 0; m < 2; ++m) {
        int row = m * 16 + r + mhalf * 8;
        int t = tok_s[row];
        if (t >= 0)
          out[(size_t)t * HIDDEN + col] = gate_s[row] * (yacc[m][nt][r] + b2v);
      }
    }
  }
}

extern "C" void kernel_launch(void* const* d_in, const int* in_sizes, int n_in,
                              void* d_out, int out_size, void* d_ws, size_t ws_size,
                              hipStream_t stream) {
  const float* x  = (const float*)d_in[0];
  const float* Wg = (const float*)d_in[1];
  const float* W1 = (const float*)d_in[2];
  const float* b1 = (const float*)d_in[3];
  const float* W2 = (const float*)d_in[4];
  const float* b2 = (const float*)d_in[5];
  float* out = (float*)d_out;

  char*     ws       = (char*)d_ws;
  int*      cursor   = (int*)     (ws);
  int*      idx_buf  = (int*)     (ws + 256);
  float*    gate_buf = (float*)   (ws + 256 + 4 * NTOK);
  float*    probs    = (float*)   (ws + 256 + 8 * NTOK);
  int*      list     = (int*)     (ws + 256 + 8 * NTOK + 4 * NTOK * NEXP);
  // packed weights: 8 experts x 2048 fragments x 1KB each
  unsigned* W1pk     = (unsigned*)(ws + 256 + 8 * NTOK + 8 * NTOK * NEXP);
  unsigned* W2pk     = W1pk + (size_t)NEXP * FRAGS1 * 256;

  hipLaunchKernelGGL(init_kernel, dim3(1), dim3(32), 0, stream, cursor);
  hipLaunchKernelGGL(gate_kernel, dim3(NTOK / 4), dim3(128), 0, stream,
                     x, Wg, cursor, idx_buf, gate_buf, probs, list);
  hipLaunchKernelGGL(loss_kernel, dim3(1), dim3(256), 0, stream,
                     probs, idx_buf, out + (size_t)NTOK * HIDDEN);
  // wave per fragment, 4 waves per block: NEXP*FRAGS = 16384 waves -> 4096 blocks
  hipLaunchKernelGGL(pack_kernel, dim3(NEXP * FRAGS1 / 4), dim3(128), 0, stream,
                     W1, W1pk, HIDDEN, FFN, KS1);
  hipLaunchKernelGGL(pack_kernel, dim3(NEXP * FRAGS2 / 4), dim3(128), 0, stream,
                     W2, W2pk, FFN, HIDDEN, KS2);
  hipLaunchKernelGGL(moe_kernel, dim3(NTOK / TILE_M, NEXP), dim3(128), 0, stream,
                     x, W1pk, b1, W2pk, b2, gate_buf, list, cursor, out);
}
